// MultiHeadAttention_42631845380644
// MI455X (gfx1250) — matrix operand order repaired and checked by reading, not compiled, not run on hardware
//
#include <hip/hip_runtime.h>

#define HIDDEN   1024
#define HEADS    16
#define HEAD_DIM 64
#define BATCH    2
#define SEQ      2048
#define MROWS    (BATCH * SEQ)   // 4096
#define NQKV     (3 * HIDDEN)    // 3072

typedef __attribute__((ext_vector_type(16))) __bf16 v16bf;
typedef __attribute__((ext_vector_type(8)))  float  v8f;
typedef __attribute__((ext_vector_type(4)))  unsigned int u32x4;
typedef __attribute__((ext_vector_type(8)))  int i32x8;
typedef __attribute__((ext_vector_type(4)))  int i32x4;

union FragAB { uint4 u[2]; v16bf v; };

#define WMMA_BF16(A_, B_, C_) \
  __builtin_amdgcn_wmma_f32_16x16x32_bf16(false, (A_), false, (B_), (short)0, (C_), false, false)

static __device__ __forceinline__ unsigned short f32_to_bf16(float f) {
  union { float f; unsigned u; } x; x.f = f;
  unsigned r = x.u + 0x7FFFu + ((x.u >> 16) & 1u);
  return (unsigned short)(r >> 16);
}

static __device__ __forceinline__ v8f zero8() {
  v8f z = {0.f, 0.f, 0.f, 0.f, 0.f, 0.f, 0.f, 0.f};
  return z;
}

// --------------------------------------------------------------------------
// TDM: DMA a 2-D bf16 tile [tile_rows x tile_cols] from global to LDS.
// D# packing per CDNA5 ISA ch.8: group0 = count/lds_addr/global_addr/type,
// group1 = data_size, tensor dims, tile dims, dim0 stride. Groups 2/3 zero
// (2-D tensor). Issued by one wave; completion tracked with TENSORcnt.
// --------------------------------------------------------------------------
static __device__ __forceinline__ void tdm_load_tile_2d(
    unsigned lds_off, const void* gptr,
    unsigned tile_cols, unsigned tile_rows,
    unsigned tensor_cols, unsigned tensor_rows,
    unsigned row_stride_elems) {
  unsigned long long ga = (unsigned long long)(size_t)gptr;
  u32x4 g0;
  g0[0] = 1u;                                                  // count=1, user mode
  g0[1] = lds_off;                                             // lds_addr (bytes)
  g0[2] = (unsigned)ga;                                        // global_addr[31:0]
  g0[3] = (unsigned)((ga >> 32) & 0x01FFFFFFu) | (2u << 30);   // addr[56:32] | type=2
  i32x8 g1;
  g1[0] = 0x00010000;                                          // data_size=1 -> 2 bytes
  g1[1] = (int)((tensor_cols & 0xFFFFu) << 16);                // tensor_dim0[15:0]
  g1[2] = (int)(((tensor_cols >> 16) & 0xFFFFu) |
                ((tensor_rows & 0xFFFFu) << 16));              // dim0 hi | dim1 lo
  g1[3] = (int)(((tensor_rows >> 16) & 0xFFFFu) |
                ((tile_cols & 0xFFFFu) << 16));                // dim1 hi | tile_dim0
  g1[4] = (int)(tile_rows & 0xFFFFu);                          // tile_dim1 (tile_dim2=0)
  g1[5] = (int)row_stride_elems;                               // tensor_dim0_stride lo
  g1[6] = 0;                                                   // stride hi | dim1_stride lo
  g1[7] = 0;
  i32x4 gz = {0, 0, 0, 0};
#if defined(__clang_major__) && (__clang_major__ >= 23)
  i32x8 gz8 = {0, 0, 0, 0, 0, 0, 0, 0};
  __builtin_amdgcn_tensor_load_to_lds(g0, g1, gz, gz, gz8, 0);
#else
  __builtin_amdgcn_tensor_load_to_lds(g0, g1, gz, gz, 0);
#endif
}

static __device__ __forceinline__ unsigned lds_byte_off(const void* p) {
  return (unsigned)(size_t)p;   // LDS generic address: low 32 bits = LDS offset
}

// ---------------------------------------------------------------- convert f32 -> bf16
__global__ __launch_bounds__(256) void cvt_bf16_kernel(const float* __restrict__ src,
                                                       unsigned short* __restrict__ dst,
                                                       int n4) {
  int i = blockIdx.x * blockDim.x + threadIdx.x;
  if (i < n4) {
    float4 f = ((const float4*)src)[i];
    ushort4 o;
    o.x = f32_to_bf16(f.x);
    o.y = f32_to_bf16(f.y);
    o.z = f32_to_bf16(f.z);
    o.w = f32_to_bf16(f.w);
    ((ushort4*)dst)[i] = o;
  }
}

// ---------------------------------------------------------------- QKV projection GEMM
static __device__ __forceinline__ void store_qkv_tile(
    v8f acc, int mt, int nt, int lane,
    const float* __restrict__ bias,
    unsigned short* __restrict__ q_ws,
    unsigned short* __restrict__ k_ws,
    unsigned short* __restrict__ vT_ws) {
  const int half = lane >> 4;
  const int col  = lane & 15;
#pragma unroll
  for (int r = 0; r < 8; ++r) {
    int m = mt + r + half * 8;
    int e = nt + col;
    float v = acc[r] + bias[e];
    int b  = m >> 11;          // / SEQ
    int s  = m & (SEQ - 1);
    int h  = e / (3 * HEAD_DIM);
    int rr = e - h * (3 * HEAD_DIM);
    size_t bh = (size_t)(b * HEADS + h);
    if (rr < HEAD_DIM) {
      q_ws[(bh * SEQ + s) * HEAD_DIM + rr] = f32_to_bf16(v * 0.125f);  // fold 1/sqrt(Dh)
    } else if (rr < 2 * HEAD_DIM) {
      k_ws[(bh * SEQ + s) * HEAD_DIM + (rr - HEAD_DIM)] = f32_to_bf16(v);
    } else {
      vT_ws[(bh * HEAD_DIM + (rr - 2 * HEAD_DIM)) * SEQ + s] = f32_to_bf16(v);
    }
  }
}

// Block tile 128(M) x 64(N); A/B tiles staged through LDS by TDM, double-buffered.
__global__ __launch_bounds__(256) void gemm_qkv_kernel(
    const unsigned short* __restrict__ A,   // xb   [4096,1024] bf16
    const unsigned short* __restrict__ W,   // wqb  [3072,1024] bf16
    const float* __restrict__ bias,         // qkv_b [3072]
    unsigned short* __restrict__ q_ws,
    unsigned short* __restrict__ k_ws,
    unsigned short* __restrict__ vT_ws) {
  __shared__ __align__(16) unsigned short ldsA[2][128 * 32];
  __shared__ __align__(16) unsigned short ldsB[2][64 * 32];

  const int lane = threadIdx.x & 31;
  const int wave = threadIdx.x >> 5;
  const int nbn  = NQKV / 64;                 // 48
  const int bm   = blockIdx.x / nbn;
  const int bn   = blockIdx.x % nbn;
  const int row  = lane & 15;
  const int kbA  = (lane < 16) ? 0 : 8;
  const int kbB  = (lane < 16) ? 0 : 8;
  const int mloc = (wave & 3) * 32;           // wave's M offset in block tile
  const int nloc = (wave >> 2) * 32;          // wave's N offset in block tile

  const unsigned short* Ablk = A + (size_t)(bm * 128) * HIDDEN;
  const unsigned short* Bblk = W + (size_t)(bn * 64) * HIDDEN;

  v8f acc00 = zero8(), acc01 = zero8(), acc10 = zero8(), acc11 = zero8();

  if (wave == 0) {
    tdm_load_tile_2d(lds_byte_off(&ldsA[0][0]), Ablk, 32, 128, HIDDEN, MROWS, HIDDEN);
    tdm_load_tile_2d(lds_byte_off(&ldsB[0][0]), Bblk, 32, 64, HIDDEN, NQKV, HIDDEN);
  }

  int p = 0;
  for (int k0 = 0; k0 < HIDDEN; k0 += 32) {
    if (wave == 0) {
      if (k0 + 32 < HIDDEN) {
        tdm_load_tile_2d(lds_byte_off(&ldsA[p ^ 1][0]), Ablk + k0 + 32, 32, 128,
                         HIDDEN, MROWS, HIDDEN);
        tdm_load_tile_2d(lds_byte_off(&ldsB[p ^ 1][0]), Bblk + k0 + 32, 32, 64,
                         HIDDEN, NQKV, HIDDEN);
        __builtin_amdgcn_s_wait_tensorcnt((short)2);   // buffer p complete (in-order)
      } else {
        __builtin_amdgcn_s_wait_tensorcnt((short)0);
      }
    }
    __syncthreads();   // tile p visible to all waves

    const unsigned short* lA = &ldsA[p][0];
    const unsigned short* lB = &ldsB[p][0];
    FragAB a0, a1, b0, b1;
    a0.u[0] = *(const uint4*)(lA + (mloc + row) * 32 + kbA);
    a0.u[1] = *(const uint4*)(lA + (mloc + row) * 32 + kbA + 16);
    a1.u[0] = *(const uint4*)(lA + (mloc + 16 + row) * 32 + kbA);
    a1.u[1] = *(const uint4*)(lA + (mloc + 16 + row) * 32 + kbA + 16);
    b0.u[0] = *(const uint4*)(lB + (nloc + row) * 32 + kbB);
    b0.u[1] = *(const uint4*)(lB + (nloc + row) * 32 + kbB + 16);
    b1.u[0] = *(const uint4*)(lB + (nloc + 16 + row) * 32 + kbB);
    b1.u[1] = *(const uint4*)(lB + (nloc + 16 + row) * 32 + kbB + 16);
    acc00 = WMMA_BF16(a0.v, b0.v, acc00);
    acc01 = WMMA_BF16(a0.v, b1.v, acc01);
    acc10 = WMMA_BF16(a1.v, b0.v, acc10);
    acc11 = WMMA_BF16(a1.v, b1.v, acc11);

    __syncthreads();   // reads of buffer p done before it is overwritten
    p ^= 1;
  }

  const int m0 = bm * 128 + mloc;
  const int n0 = bn * 64 + nloc;
  store_qkv_tile(acc00, m0,      n0,      lane, bias, q_ws, k_ws, vT_ws);
  store_qkv_tile(acc01, m0,      n0 + 16, lane, bias, q_ws, k_ws, vT_ws);
  store_qkv_tile(acc10, m0 + 16, n0,      lane, bias, q_ws, k_ws, vT_ws);
  store_qkv_tile(acc11, m0 + 16, n0 + 16, lane, bias, q_ws, k_ws, vT_ws);
}

// ---------------------------------------------------------------- flash attention
__global__ __launch_bounds__(256) void attn_kernel(
    const unsigned short* __restrict__ q_ws,   // [BH, S, 64] bf16, pre-scaled
    const unsigned short* __restrict__ k_ws,   // [BH, S, 64] bf16
    const unsigned short* __restrict__ vT_ws,  // [BH, 64, S] bf16
    unsigned short* __restrict__ attn_ws) {    // [B*S, 1024] bf16 (heads merged)
  __shared__ __align__(16) unsigned short plds[8][16][32];

  const int lane = threadIdx.x & 31;
  const int wave = threadIdx.x >> 5;
  const int tilesPerBH = SEQ / 128;            // 16
  const int bh = blockIdx.x / tilesPerBH;
  const int qt = blockIdx.x % tilesPerBH;
  const int m0 = qt * 128 + wave * 16;
  const int row  = lane & 15;
  const int half = lane >> 4;
  const int kbA  = half ? 8 : 0;
  const int kbB  = half ? 8 : 0;

  FragAB aq0, aq1;
  {
    const unsigned short* qp = q_ws + ((size_t)bh * SEQ + m0 + row) * HEAD_DIM;
    aq0.u[0] = *(const uint4*)(qp + kbA);
    aq0.u[1] = *(const uint4*)(qp + kbA + 16);
    aq1.u[0] = *(const uint4*)(qp + 32 + kbA);
    aq1.u[1] = *(const uint4*)(qp + 32 + kbA + 16);
  }

  v8f o0 = zero8(), o1 = zero8(), o2 = zero8(), o3 = zero8();
  float mrun[8], lrun[8];
#pragma unroll
  for (int r = 0; r < 8; ++r) { mrun[r] = -3.0e38f; lrun[r] = 0.f; }

  for (int n0 = 0; n0 < SEQ; n0 += 32) {
    v8f c0 = zero8(), c1 = zero8();
    {
      FragAB bk;
      const unsigned short* kp = k_ws + ((size_t)bh * SEQ + n0 + row) * HEAD_DIM;
      bk.u[0] = *(const uint4*)(kp + kbB);
      bk.u[1] = *(const uint4*)(kp + kbB + 16);
      c0 = WMMA_BF16(aq0.v, bk.v, c0);
      bk.u[0] = *(const uint4*)(kp + 32 + kbB);
      bk.u[1] = *(const uint4*)(kp + 32 + kbB + 16);
      c0 = WMMA_BF16(aq1.v, bk.v, c0);
      const unsigned short* kp1 = kp + 16 * HEAD_DIM;
      bk.u[0] = *(const uint4*)(kp1 + kbB);
      bk.u[1] = *(const uint4*)(kp1 + kbB + 16);
      c1 = WMMA_BF16(aq0.v, bk.v, c1);
      bk.u[0] = *(const uint4*)(kp1 + 32 + kbB);
      bk.u[1] = *(const uint4*)(kp1 + 32 + kbB + 16);
      c1 = WMMA_BF16(aq1.v, bk.v, c1);
    }

    // online softmax; rows m = r + 8*half live in 16-lane halves
    float pm[8];
#pragma unroll
    for (int r = 0; r < 8; ++r) pm[r] = fmaxf(c0[r], c1[r]);
#pragma unroll
    for (int mask = 1; mask <= 8; mask <<= 1) {
#pragma unroll
      for (int r = 0; r < 8; ++r) pm[r] = fmaxf(pm[r], __shfl_xor(pm[r], mask, 32));
    }
    float alpha[8], p0[8], p1[8], psum[8];
#pragma unroll
    for (int r = 0; r < 8; ++r) {
      float nm = fmaxf(mrun[r], pm[r]);
      alpha[r] = __expf(mrun[r] - nm);
      mrun[r]  = nm;
      p0[r] = __expf(c0[r] - nm);
      p1[r] = __expf(c1[r] - nm);
      psum[r] = p0[r] + p1[r];
    }
#pragma unroll
    for (int mask = 1; mask <= 8; mask <<= 1) {
#pragma unroll
      for (int r = 0; r < 8; ++r) psum[r] += __shfl_xor(psum[r], mask, 32);
    }
#pragma unroll
    for (int r = 0; r < 8; ++r) {
      lrun[r] = lrun[r] * alpha[r] + psum[r];
      o0[r] *= alpha[r]; o1[r] *= alpha[r]; o2[r] *= alpha[r]; o3[r] *= alpha[r];
    }

    // C-layout -> A-layout transpose of P through per-wave LDS
#pragma unroll
    for (int r = 0; r < 8; ++r) {
      int rm = r + half * 8;
      plds[wave][rm][row]      = f32_to_bf16(p0[r]);
      plds[wave][rm][16 + row] = f32_to_bf16(p1[r]);
    }
    asm volatile("s_wait_dscnt 0" ::: "memory");
    FragAB ap;
    ap.u[0] = *(const uint4*)(&plds[wave][row][kbA]);
    ap.u[1] = *(const uint4*)(&plds[wave][row][kbA + 16]);

    // PV: O[16x64] += P[16x32] * V[32x64]
    {
      FragAB bv;
      const unsigned short* vp = vT_ws + ((size_t)bh * HEAD_DIM + row) * SEQ + n0;
      bv.u[0] = *(const uint4*)(vp + kbB);
      bv.u[1] = *(const uint4*)(vp + kbB + 16);
      o0 = WMMA_BF16(ap.v, bv.v, o0);
      vp += 16 * SEQ;
      bv.u[0] = *(const uint4*)(vp + kbB);
      bv.u[1] = *(const uint4*)(vp + kbB + 16);
      o1 = WMMA_BF16(ap.v, bv.v, o1);
      vp += 16 * SEQ;
      bv.u[0] = *(const uint4*)(vp + kbB);
      bv.u[1] = *(const uint4*)(vp + kbB + 16);
      o2 = WMMA_BF16(ap.v, bv.v, o2);
      vp += 16 * SEQ;
      bv.u[0] = *(const uint4*)(vp + kbB);
      bv.u[1] = *(const uint4*)(vp + kbB + 16);
      o3 = WMMA_BF16(ap.v, bv.v, o3);
    }
  }

  const int bb = bh >> 4;   // / HEADS
  const int hh = bh & 15;
#pragma unroll
  for (int r = 0; r < 8; ++r) {
    float inv = 1.f / lrun[r];
    int m = m0 + r + half * 8;
    size_t base = ((size_t)bb * SEQ + m) * HIDDEN + hh * HEAD_DIM;
    attn_ws[base +      row] = f32_to_bf16(o0[r] * inv);
    attn_ws[base + 16 + row] = f32_to_bf16(o1[r] * inv);
    attn_ws[base + 32 + row] = f32_to_bf16(o2[r] * inv);
    attn_ws[base + 48 + row] = f32_to_bf16(o3[r] * inv);
  }
}

// ---------------------------------------------------------------- output projection
__global__ __launch_bounds__(256) void gemm_out_kernel(
    const unsigned short* __restrict__ A,   // attn_ws [4096,1024] bf16
    const unsigned short* __restrict__ W,   // wob [1024,1024] bf16
    const float* __restrict__ bias,         // o_b [1024]
    float* __restrict__ out) {              // [4096,1024] f32
  __shared__ __align__(16) unsigned short ldsA[2][128 * 32];
  __shared__ __align__(16) unsigned short ldsB[2][64 * 32];

  const int lane = threadIdx.x & 31;
  const int wave = threadIdx.x >> 5;
  const int nbn  = HIDDEN / 64;               // 16
  const int bm   = blockIdx.x / nbn;
  const int bn   = blockIdx.x % nbn;
  const int row  = lane & 15;
  const int kbA  = (lane < 16) ? 0 : 8;
  const int kbB  = (lane < 16) ? 0 : 8;
  const int mloc = (wave & 3) * 32;
  const int nloc = (wave >> 2) * 32;

  const unsigned short* Ablk = A + (size_t)(bm * 128) * HIDDEN;
  const unsigned short* Bblk = W + (size_t)(bn * 64) * HIDDEN;

  v8f acc00 = zero8(), acc01 = zero8(), acc10 = zero8(), acc11 = zero8();

  if (wave == 0) {
    tdm_load_tile_2d(lds_byte_off(&ldsA[0][0]), Ablk, 32, 128, HIDDEN, MROWS, HIDDEN);
    tdm_load_tile_2d(lds_byte_off(&ldsB[0][0]), Bblk, 32, 64, HIDDEN, HIDDEN, HIDDEN);
  }

  int p = 0;
  for (int k0 = 0; k0 < HIDDEN; k0 += 32) {
    if (wave == 0) {
      if (k0 + 32 < HIDDEN) {
        tdm_load_tile_2d(lds_byte_off(&ldsA[p ^ 1][0]), Ablk + k0 + 32, 32, 128,
                         HIDDEN, MROWS, HIDDEN);
        tdm_load_tile_2d(lds_byte_off(&ldsB[p ^ 1][0]), Bblk + k0 + 32, 32, 64,
                         HIDDEN, HIDDEN, HIDDEN);
        __builtin_amdgcn_s_wait_tensorcnt((short)2);
      } else {
        __builtin_amdgcn_s_wait_tensorcnt((short)0);
      }
    }
    __syncthreads();

    const unsigned short* lA = &ldsA[p][0];
    const unsigned short* lB = &ldsB[p][0];
    FragAB a0, a1, b0, b1;
    a0.u[0] = *(const uint4*)(lA + (mloc + row) * 32 + kbA);
    a0.u[1] = *(const uint4*)(lA + (mloc + row) * 32 + kbA + 16);
    a1.u[0] = *(const uint4*)(lA + (mloc + 16 + row) * 32 + kbA);
    a1.u[1] = *(const uint4*)(lA + (mloc + 16 + row) * 32 + kbA + 16);
    b0.u[0] = *(const uint4*)(lB + (nloc + row) * 32 + kbB);
    b0.u[1] = *(const uint4*)(lB + (nloc + row) * 32 + kbB + 16);
    b1.u[0] = *(const uint4*)(lB + (nloc + 16 + row) * 32 + kbB);
    b1.u[1] = *(const uint4*)(lB + (nloc + 16 + row) * 32 + kbB + 16);
    acc00 = WMMA_BF16(a0.v, b0.v, acc00);
    acc01 = WMMA_BF16(a0.v, b1.v, acc01);
    acc10 = WMMA_BF16(a1.v, b0.v, acc10);
    acc11 = WMMA_BF16(a1.v, b1.v, acc11);

    __syncthreads();
    p ^= 1;
  }

  const int m0 = bm * 128 + mloc;
  const int n0 = bn * 64 + nloc;
  const int half = lane >> 4;
  const int col  = lane & 15;
#pragma unroll
  for (int r = 0; r < 8; ++r) {
    int mA = m0 + r + half * 8;
    int mB = mA + 16;
    out[(size_t)mA * HIDDEN + n0 + col]      = acc00[r] + bias[n0 + col];
    out[(size_t)mA * HIDDEN + n0 + 16 + col] = acc01[r] + bias[n0 + 16 + col];
    out[(size_t)mB * HIDDEN + n0 + col]      = acc10[r] + bias[n0 + col];
    out[(size_t)mB * HIDDEN + n0 + 16 + col] = acc11[r] + bias[n0 + 16 + col];
  }
}

// ---------------------------------------------------------------- launch
extern "C" void kernel_launch(void* const* d_in, const int* in_sizes, int n_in,
                              void* d_out, int out_size, void* d_ws, size_t ws_size,
                              hipStream_t stream) {
  const float* x     = (const float*)d_in[0];  // [2,2048,1024]
  const float* qkv_w = (const float*)d_in[1];  // [3072,1024]
  const float* qkv_b = (const float*)d_in[2];  // [3072]
  const float* o_w   = (const float*)d_in[3];  // [1024,1024]
  const float* o_b   = (const float*)d_in[4];  // [1024]
  float* out = (float*)d_out;

  char* ws = (char*)d_ws;
  size_t off = 0;
  auto carve = [&](size_t bytes) -> void* {
    void* p = ws + off;
    off = (off + bytes + 255) & ~(size_t)255;
    return p;
  };
  unsigned short* xb      = (unsigned short*)carve((size_t)MROWS * HIDDEN * 2);
  unsigned short* wqb     = (unsigned short*)carve((size_t)NQKV * HIDDEN * 2);
  unsigned short* wob     = (unsigned short*)carve((size_t)HIDDEN * HIDDEN * 2);
  unsigned short* q_ws    = (unsigned short*)carve((size_t)BATCH * HEADS * SEQ * HEAD_DIM * 2);
  unsigned short* k_ws    = (unsigned short*)carve((size_t)BATCH * HEADS * SEQ * HEAD_DIM * 2);
  unsigned short* vT_ws   = (unsigned short*)carve((size_t)BATCH * HEADS * SEQ * HEAD_DIM * 2);
  unsigned short* attn_ws = (unsigned short*)carve((size_t)MROWS * HIDDEN * 2);

  cvt_bf16_kernel<<<(MROWS * HIDDEN / 4 + 255) / 256, 256, 0, stream>>>(x, xb, MROWS * HIDDEN / 4);
  cvt_bf16_kernel<<<(NQKV * HIDDEN / 4 + 255) / 256, 256, 0, stream>>>(qkv_w, wqb, NQKV * HIDDEN / 4);
  cvt_bf16_kernel<<<(HIDDEN * HIDDEN / 4 + 255) / 256, 256, 0, stream>>>(o_w, wob, HIDDEN * HIDDEN / 4);

  gemm_qkv_kernel<<<(MROWS / 128) * (NQKV / 64), 256, 0, stream>>>(xb, wqb, qkv_b, q_ws, k_ws, vT_ws);
  attn_kernel<<<BATCH * HEADS * (SEQ / 128), 256, 0, stream>>>(q_ws, k_ws, vT_ws, attn_ws);
  gemm_out_kernel<<<(MROWS / 128) * (HIDDEN / 64), 256, 0, stream>>>(attn_ws, wob, o_b, out);
}
